// AttentionGrapher_63385127354705
// MI455X (gfx1250) — matrix operand order repaired and checked by reading, not compiled, not run on hardware
//
#include <hip/hip_runtime.h>

// Problem constants (B, C, H, W, K) = (4, 96, 56, 56, 9), CM = 192
#define BB   4
#define CC   96
#define CMC  192
#define HHH  56
#define WWW  56
#define NNN  3136     // 56*56
#define KNN  9

typedef __attribute__((ext_vector_type(16))) __bf16          v16bf;
typedef __attribute__((ext_vector_type(8)))  float           v8f;
typedef __attribute__((ext_vector_type(8)))  unsigned short  v8us;

__device__ __forceinline__ unsigned short f2bf(float f) {
  // round-to-nearest-even float -> bf16 bits
  unsigned int u = __float_as_uint(f);
  unsigned int r = u + 0x7FFFu + ((u >> 16) & 1u);
  return (unsigned short)(r >> 16);
}

// ---------------------------------------------------------------------------
// Weight prep: fc1_w -> bf16 ; gconv_w -> W12 = [Wl - Wr ; Wr] bf16 ; fc2_w bf16
// ---------------------------------------------------------------------------
__global__ void prep_weights_kernel(const float* __restrict__ fc1_w,
                                    const float* __restrict__ gconv_w,
                                    const float* __restrict__ fc2_w,
                                    unsigned short* __restrict__ fc1wb,
                                    unsigned short* __restrict__ w12b,
                                    unsigned short* __restrict__ fc2wb) {
  int t = blockIdx.x * blockDim.x + threadIdx.x;
  if (t < CC * CC) { fc1wb[t] = f2bf(fc1_w[t]); return; }
  t -= CC * CC;
  if (t < 2 * CMC * CC) {
    int o = t / CC, c = t % CC;
    float v;
    if (o < CMC) v = gconv_w[o * (2 * CC) + c] - gconv_w[o * (2 * CC) + CC + c];
    else         v = gconv_w[(o - CMC) * (2 * CC) + CC + c];
    w12b[t] = f2bf(v);
    return;
  }
  t -= 2 * CMC * CC;
  if (t < CC * CMC) fc2wb[t] = f2bf(fc2_w[t]);
}

// ---------------------------------------------------------------------------
// LDS-tiled transpose + convert: x[b][c][n] f32  ->  x_t[b][n][c] bf16
// ---------------------------------------------------------------------------
__global__ __launch_bounds__(256)
void transpose_convert_kernel(const float* __restrict__ src,
                              unsigned short* __restrict__ dst) {
  __shared__ float tile[32][33];
  const int n0 = blockIdx.x * 32;
  const int c0 = blockIdx.y * 32;
  const int b  = blockIdx.z;
  const int tx = threadIdx.x, ty = threadIdx.y;
#pragma unroll
  for (int r = 0; r < 4; ++r) {
    int c = c0 + ty + r * 8;
    tile[ty + r * 8][tx] = src[((size_t)b * CC + c) * NNN + n0 + tx];
  }
  __syncthreads();
#pragma unroll
  for (int r = 0; r < 4; ++r) {
    int nl = ty + r * 8;
    dst[((size_t)b * NNN + n0 + nl) * CC + c0 + tx] = f2bf(tile[tx][nl]);
  }
}

// ---------------------------------------------------------------------------
// Generic batched GEMM via v_wmma_f32_16x16x32_bf16.
//   D[m][n] = sum_k W[m][k] * Xt[b][n][k] (+ bias[m])
// W:  [M][KD] row-major bf16 (K contiguous)
// Xt: [b][Nn][KD] bf16 (K contiguous)  -> B fragment = 2x b128 per lane
// Compile-time output modes (branch-free specializations):
//   OF  : f32  [b][M][Nn]  (standard layout, for d_out)
//   OT  : f32  [b][Nn][M]  (K-contiguous layout, vector stores)
//   OTB : bf16 [b][Nn][M]  (K-contiguous layout, vector stores)
// One wave per 16x16 tile; block = 8 waves over 8 M-tiles; K fully unrolled.
// ---------------------------------------------------------------------------
template <int KD, bool BIAS, bool OF, bool OT, bool OTB>
__global__ __launch_bounds__(256)
void gemm_bf16_wmma_kernel(const unsigned short* __restrict__ Wp,
                           const unsigned short* __restrict__ Xp,
                           const float*  __restrict__ bias,
                           float* __restrict__ outF,
                           float* __restrict__ outT,
                           unsigned short* __restrict__ outTB,
                           int M, int Nn) {
  const int mtile = blockIdx.y * blockDim.y + threadIdx.y;
  if (mtile * 16 >= M) return;
  const int b    = blockIdx.z;
  const int lane = threadIdx.x;       // 0..31 (wave32)
  const int row  = lane & 15;
  const int half = lane >> 4;
  const int col  = blockIdx.x * 16 + row;
  const unsigned short* Xb = Xp + ((size_t)b * Nn + col) * KD;
  const unsigned short* Wr = Wp + (size_t)(mtile * 16 + row) * KD;
  v8f acc = {};
#pragma unroll
  for (int kb = 0; kb < KD; kb += 32) {
    // A: lane<16 holds K {kb..kb+7, kb+16..kb+23}; lane>=16 shifted by 8
    const v8us* ap = (const v8us*)(Wr + kb + 8 * half);
    v8us a0 = ap[0], a1 = ap[2];
    v16bf av = __builtin_bit_cast(v16bf, __builtin_shufflevector(
        a0, a1, 0, 1, 2, 3, 4, 5, 6, 7, 8, 9, 10, 11, 12, 13, 14, 15));
    // B: lane holds column col, K {kb..kb+7, kb+16..kb+23}; lane>=16 shifted by 8
    const v8us* bp = (const v8us*)(Xb + kb + 8 * half);
    v8us b0 = bp[0], b1 = bp[2];
    v16bf bv = __builtin_bit_cast(v16bf, __builtin_shufflevector(
        b0, b1, 0, 1, 2, 3, 4, 5, 6, 7, 8, 9, 10, 11, 12, 13, 14, 15));
    acc = __builtin_amdgcn_wmma_f32_16x16x32_bf16(false, av, false, bv,
                                                  (short)0, acc, false, false);
  }
  // D layout: VGPR r of lane -> row (mtile*16 + 8*half + r), column col
  const int m0 = mtile * 16 + 8 * half;
  v8f vals = acc;
  if (BIAS) {
    v8f bvec = *(const v8f*)(bias + m0);   // 8 consecutive floats, 32B aligned
#pragma unroll
    for (int r = 0; r < 8; ++r) vals[r] += bvec[r];
  }
  if (OF) {
    size_t o = (size_t)b * M * Nn + (size_t)m0 * Nn + col;
#pragma unroll
    for (int r = 0; r < 8; ++r) outF[o + (size_t)r * Nn] = vals[r];
  }
  if (OT) {   // 8 consecutive floats -> vector store
    *(v8f*)(outT + ((size_t)b * Nn + col) * M + m0) = vals;
  }
  if (OTB) {  // 8 consecutive bf16 -> single b128 store
    v8us bfv;
#pragma unroll
    for (int r = 0; r < 8; ++r) bfv[r] = f2bf(vals[r]);
    *(v8us*)(outTB + ((size_t)b * Nn + col) * M + m0) = bfv;
  }
}

// ---------------------------------------------------------------------------
// Row squared norms: x2[b][n] = sum_c h_t[b][n][c]^2   (contiguous rows)
// ---------------------------------------------------------------------------
__global__ void rownorm_kernel(const float* __restrict__ h_t, float* __restrict__ x2) {
  int t = blockIdx.x * blockDim.x + threadIdx.x;
  if (t >= BB * NNN) return;
  const float4* p = (const float4*)(h_t + (size_t)t * CC);
  float s = 0.f;
#pragma unroll 6
  for (int i = 0; i < CC / 4; ++i) {
    float4 v = p[i];
    s += v.x * v.x + v.y * v.y + v.z * v.z + v.w * v.w;
  }
  x2[t] = s;
}

// ---------------------------------------------------------------------------
// KNN: for each query n, top-9 smallest (x2[m] - 2<h_n, h_m>).
// h_t layout [b][n][96]: candidate chunk = one contiguous 12KB block -> LDS.
// Block = 8 waves = 8 queries; per-lane sorted top-9 then wave shuffle-merge.
// ---------------------------------------------------------------------------
__global__ __launch_bounds__(256)
void knn_kernel(const float* __restrict__ h_t, const float* __restrict__ x2,
                int* __restrict__ idx) {
  __shared__ float qf[8][CC];
  __shared__ float cand[CC][33];
  const int b    = blockIdx.y;
  const int n0   = blockIdx.x * 8;
  const int lane = threadIdx.x;
  const int wv   = threadIdx.y;
  const int tid  = wv * 32 + lane;
  const float* hb  = h_t + (size_t)b * NNN * CC;
  const float* x2b = x2 + (size_t)b * NNN;
  for (int e = tid; e < 8 * CC; e += 256) {
    int q = e / CC, c = e % CC;
    qf[q][c] = hb[(size_t)(n0 + q) * CC + c];
  }
  float d9[KNN]; int i9[KNN];
#pragma unroll
  for (int j = 0; j < KNN; ++j) { d9[j] = 3.0e38f; i9[j] = 0x7FFFFFFF; }
  for (int m0 = 0; m0 < NNN; m0 += 32) {
    __syncthreads();
    for (int e = tid; e < 32 * CC; e += 256) {
      int mm = e / CC, c = e % CC;                    // coalesced reads
      cand[c][mm] = hb[(size_t)(m0 + mm) * CC + c];   // padded LDS, no conflict
    }
    __syncthreads();
    float dot = 0.f;
#pragma unroll 8
    for (int c = 0; c < CC; ++c) dot = fmaf(qf[wv][c], cand[c][lane], dot);
    int m = m0 + lane;
    float score = x2b[m] - 2.0f * dot;
    if (score < d9[KNN - 1]) {           // sorted insertion, stable on ties
      int j = KNN - 1;
      while (j > 0 && d9[j - 1] > score) { d9[j] = d9[j - 1]; i9[j] = i9[j - 1]; --j; }
      d9[j] = score; i9[j] = m;
    }
  }
  // Wave-wide 9-way selection merge of 32 sorted lists.
  int p = 0;
  int* out = idx + ((size_t)b * NNN + (n0 + wv)) * KNN;
#pragma unroll
  for (int s = 0; s < KNN; ++s) {
    float v  = (p < KNN) ? d9[p] : 3.0e38f;
    int   mi = (p < KNN) ? i9[p] : 0x7FFFFFFF;
    float bvv = v; int bii = mi;
#pragma unroll
    for (int off = 16; off > 0; off >>= 1) {
      float ov = __shfl_xor(bvv, off, 32);
      int   oi = __shfl_xor(bii, off, 32);
      if (ov < bvv || (ov == bvv && oi < bii)) { bvv = ov; bii = oi; }
    }
    if (p < KNN && mi == bii) ++p;       // unique indices -> single winner
    if (lane == 0) out[s] = bii;
  }
}

// ---------------------------------------------------------------------------
// g_t[b][n][o] = relu( max_k ( A[n][o] + bias[o] + Bv[idx[n][k]][o] ) ), bf16.
// AB_t layout [b][n][384]: cols 0..191 = (Wl-Wr)h, cols 192..383 = Wr h.
// One block (192 threads = 6 waves) per pixel; channels across threads.
// ---------------------------------------------------------------------------
__global__ __launch_bounds__(192)
void gather_max_kernel(const float* __restrict__ AB_t,
                       const float* __restrict__ gconv_b,
                       const int* __restrict__ idx,
                       unsigned short* __restrict__ g_t) {
  __shared__ int sidx[KNN];
  const int bn = blockIdx.x;            // b*NNN + n
  const int b  = bn / NNN;
  const int o  = threadIdx.x;           // 0..191
  const int* ip = idx + (size_t)bn * KNN;
  if (o < KNN) sidx[o] = ip[o];
  __syncthreads();
  const float* Ab = AB_t + (size_t)b * NNN * (2 * CMC);
  float a = AB_t[(size_t)bn * (2 * CMC) + o] + gconv_b[o];
  float mx = -3.0e38f;
#pragma unroll
  for (int k = 0; k < KNN; ++k)
    mx = fmaxf(mx, a + Ab[(size_t)sidx[k] * (2 * CMC) + CMC + o]);
  g_t[(size_t)bn * CMC + o] = f2bf(fmaxf(mx, 0.0f));
}

// ---------------------------------------------------------------------------
// Spatial attention: channel mean/max over d_out -> sa_in[b][2][n]
// ---------------------------------------------------------------------------
__global__ void sa_reduce_kernel(const float* __restrict__ outF,
                                 float* __restrict__ sa_in) {
  int t = blockIdx.x * blockDim.x + threadIdx.x;
  if (t >= BB * NNN) return;
  int b = t / NNN, n = t % NNN;
  const float* ob = outF + (size_t)b * CC * NNN + n;
  float s = 0.f, mx = -3.0e38f;
#pragma unroll 4
  for (int c = 0; c < CC; ++c) { float v = ob[(size_t)c * NNN]; s += v; mx = fmaxf(mx, v); }
  sa_in[((size_t)b * 2 + 0) * NNN + n] = s * (1.0f / CC);
  sa_in[((size_t)b * 2 + 1) * NNN + n] = mx;
}

// 7x7 conv (pad 3) + sigmoid, then out = out*att + shortcut (in place on d_out)
__global__ void sa_apply_kernel(float* __restrict__ outF,
                                const float* __restrict__ x,
                                const float* __restrict__ sa_in,
                                const float* __restrict__ sa_w) {
  int t = blockIdx.x * blockDim.x + threadIdx.x;
  if (t >= BB * NNN) return;
  int b = t / NNN, n = t % NNN;
  int y = n / WWW, xx = n % WWW;
  const float* sb = sa_in + (size_t)b * 2 * NNN;
  float acc = 0.f;
#pragma unroll
  for (int c = 0; c < 2; ++c)
#pragma unroll
    for (int dy = 0; dy < 7; ++dy) {
      int yy = y + dy - 3;
      if (yy < 0 || yy >= HHH) continue;
#pragma unroll
      for (int dx = 0; dx < 7; ++dx) {
        int xs = xx + dx - 3;
        if (xs < 0 || xs >= WWW) continue;
        acc = fmaf(sa_w[(c * 7 + dy) * 7 + dx], sb[(size_t)c * NNN + yy * WWW + xs], acc);
      }
    }
  float att = 1.0f / (1.0f + __expf(-acc));
  float* ob = outF + (size_t)b * CC * NNN + n;
  const float* xb = x + (size_t)b * CC * NNN + n;
  for (int c = 0; c < CC; ++c) {
    size_t o = (size_t)c * NNN;
    ob[o] = ob[o] * att + xb[o];
  }
}

// ---------------------------------------------------------------------------
extern "C" void kernel_launch(void* const* d_in, const int* in_sizes, int n_in,
                              void* d_out, int out_size, void* d_ws, size_t ws_size,
                              hipStream_t stream) {
  const float* x       = (const float*)d_in[0];
  const float* fc1_w   = (const float*)d_in[1];
  const float* fc1_b   = (const float*)d_in[2];
  const float* gconv_w = (const float*)d_in[3];
  const float* gconv_b = (const float*)d_in[4];
  const float* fc2_w   = (const float*)d_in[5];
  const float* fc2_b   = (const float*)d_in[6];
  const float* sa_w    = (const float*)d_in[7];

  char* ws = (char*)d_ws;
  size_t off = 0;
  auto alloc = [&](size_t bytes) -> void* {
    void* p = ws + off;
    off = (off + bytes + 255) & ~(size_t)255;
    return p;
  };
  unsigned short* fc1wb = (unsigned short*)alloc((size_t)CC * CC * 2);
  unsigned short* w12b  = (unsigned short*)alloc((size_t)2 * CMC * CC * 2);
  unsigned short* fc2wb = (unsigned short*)alloc((size_t)CC * CMC * 2);
  unsigned short* x_t   = (unsigned short*)alloc((size_t)BB * NNN * CC * 2);  // bf16 [b][n][c]
  float*          h_t   = (float*)alloc((size_t)BB * NNN * CC * 4);           // f32  [b][n][c]
  unsigned short* h_tb  = (unsigned short*)alloc((size_t)BB * NNN * CC * 2);  // bf16 [b][n][c]
  float*          x2    = (float*)alloc((size_t)BB * NNN * 4);
  int*            idx   = (int*)alloc((size_t)BB * NNN * KNN * 4);
  float*          AB_t  = (float*)alloc((size_t)BB * NNN * 2 * CMC * 4);      // f32  [b][n][384]
  unsigned short* g_t   = (unsigned short*)alloc((size_t)BB * NNN * CMC * 2); // bf16 [b][n][192]
  float*          sa_in = (float*)alloc((size_t)BB * 2 * NNN * 4);
  float*          outF  = (float*)d_out;

  // 1. weight prep (bf16 conversion + Wl-Wr / Wr split)
  {
    int total = CC * CC + 2 * CMC * CC + CC * CMC;
    prep_weights_kernel<<<(total + 255) / 256, 256, 0, stream>>>(
        fc1_w, gconv_w, fc2_w, fc1wb, w12b, fc2wb);
  }
  // 2. x -> bf16, transposed to [b][n][c]
  transpose_convert_kernel<<<dim3(NNN / 32, CC / 32, BB), dim3(32, 8), 0, stream>>>(x, x_t);
  // 3. fc1: h = fc1_w * x + fc1_b  (WMMA) -> h_t f32 + bf16
  gemm_bf16_wmma_kernel<CC, true, false, true, true>
      <<<dim3(NNN / 16, 1, BB), dim3(32, 8), 0, stream>>>(
      fc1wb, x_t, fc1_b, nullptr, h_t, h_tb, CC, NNN);
  // 4. squared norms
  rownorm_kernel<<<(BB * NNN + 255) / 256, 256, 0, stream>>>(h_t, x2);
  // 5. KNN top-9
  knn_kernel<<<dim3(NNN / 8, BB), dim3(32, 8), 0, stream>>>(h_t, x2, idx);
  // 6. AB = [Wl-Wr ; Wr] * h  (WMMA), M=384 -> AB_t
  gemm_bf16_wmma_kernel<CC, false, false, true, false>
      <<<dim3(NNN / 16, 3, BB), dim3(32, 8), 0, stream>>>(
      w12b, h_tb, nullptr, nullptr, AB_t, nullptr, 2 * CMC, NNN);
  // 7. gather + max + relu -> g_t (bf16)
  gather_max_kernel<<<BB * NNN, CMC, 0, stream>>>(AB_t, gconv_b, idx, g_t);
  // 8. fc2: out = fc2_w * g + fc2_b (WMMA) -> d_out (standard [b][c][n])
  gemm_bf16_wmma_kernel<CMC, true, true, false, false>
      <<<dim3(NNN / 16, 1, BB), dim3(32, 8), 0, stream>>>(
      fc2wb, g_t, fc2_b, outF, nullptr, nullptr, CC, NNN);
  // 9. channel mean/max
  sa_reduce_kernel<<<(BB * NNN + 255) / 256, 256, 0, stream>>>(outF, sa_in);
  // 10. 7x7 sigmoid attention + residual, in place
  sa_apply_kernel<<<(BB * NNN + 255) / 256, 256, 0, stream>>>(outF, x, sa_in, sa_w);
  (void)in_sizes; (void)n_in; (void)out_size; (void)ws_size;
}
